// SphericalFT_89283780150004
// MI455X (gfx1250) — matrix operand order repaired and checked by reading, not compiled, not run on hardware
//
#include <hip/hip_runtime.h>
#include <math.h>

#ifndef M_PI
#define M_PI 3.14159265358979323846
#endif

#define Bq 2
#define Nq 32768
#define Cq 16
#define L1q 33          // lmax+1
#define NSPLIT 8
#define NPTS (Nq / NSPLIT)  // 4096 points per workgroup

typedef __attribute__((ext_vector_type(8)))  _Float16 v8h;
typedef __attribute__((ext_vector_type(16))) _Float16 v16h;
typedef __attribute__((ext_vector_type(8)))  float    v8f;

union V16U { v16h v; _Float16 e[16]; };

__device__ __forceinline__ v16h cat8(v8h lo, v8h hi) {
  return __builtin_shufflevector(lo, hi, 0,1,2,3,4,5,6,7,8,9,10,11,12,13,14,15);
}

// One workgroup = (b, m, split). 8 waves, each owns a private LDS staging strip.
// Per 32-point chunk per wave:
//   phase 1 (lane = point): normalized assoc. Legendre recurrence -> f16 A staging
//                           (row-major [l-m][point]), area*cos/sin(m phi) -> LDS.
//   phase 2 (lane = chan x K-half): build f16 B tiles (cos & sin), 2 ds_load_b128
//                           per A tile, up to 6 v_wmma_f32_16x16x32_f16.
__global__ __launch_bounds__(256)
void sphft_main(const float* __restrict__ theta,
                const float* __restrict__ phi,
                const float* __restrict__ areas,
                const float* __restrict__ values,
                float* __restrict__ ws) {
  // red[w]: phase 1/2 -> A staging (48 rows x 32 pts f16 = 3KB);
  //         epilogue  -> 6 f32 16x16 partial tiles (6KB).
  __shared__ alignas(16) float red[8][1536];
  __shared__ float2 cmsm[8][32];
  __shared__ float cA[33], cB[33];

  const int tid   = threadIdx.x;
  const int w     = tid >> 5;
  const int lane  = tid & 31;

  const int wg    = blockIdx.x;
  const int split = wg & (NSPLIT - 1);
  const int m     = (wg / NSPLIT) % L1q;
  const int b     = wg / (NSPLIT * L1q);

  // Fully-normalized Legendre: Nbar(l,m) = coef(l,m)*P(l,m) (signs fold out).
  // Nbar(m,m) = sqrt((2m+1)/(4pi) * prod_{k=1..m}(2k-1)/(2k)) * s^m
  double prod = (2.0 * m + 1.0) / (4.0 * M_PI);
  for (int kk = 1; kk <= m; ++kk) prod *= (2.0 * kk - 1.0) / (2.0 * kk);
  const float seedc = (float)sqrt(prod);
  const float c1 = sqrtf(2.0f * m + 3.0f);   // Nbar(m+1,m) = c1 * x * Nbar(m,m)

  if (tid >= m + 2 && tid < 33) {            // upward-l recurrence coefficients
    const int l = tid;
    cA[l] = sqrtf(((2.0f * l - 1.0f) * (2.0f * l + 1.0f)) /
                  ((float)((l - m) * (l + m))));
    cB[l] = sqrtf(((2.0f * l + 1.0f) * (float)((l + m - 1) * (l - m - 1))) /
                  ((2.0f * l - 3.0f) * (float)((l + m) * (l - m))));
  }
  __syncthreads();

  const int R = 33 - m;            // number of l rows for this m
  const int T = (R + 15) >> 4;     // 1..3 A tiles (wave-uniform)

  v8f aC0 = {}, aC1 = {}, aC2 = {};
  v8f aS0 = {}, aS1 = {}, aS2 = {};

  _Float16* As16 = (_Float16*)&red[w][0];

  const size_t pbase = (size_t)b * Nq;
  const int r16   = lane & 15;
  const int khalf = lane >> 4;     // 0 or 1
  const int K0h   = khalf * 16;    // A-layout: contiguous K offset (f16 elems)
  const int Kbase = khalf * 16;    // B-layout: K offset
  const int ch    = lane & 15;     // B-layout: channel

  const int nend = split * NPTS + NPTS;
  for (int n0 = split * NPTS + w * 32; n0 < nend; n0 += 256) {
    // ---------------- phase 1: lane = point ----------------
    const int n = n0 + lane;
    const float th = theta[pbase + n];
    const float ph = phi[pbase + n];
    const float ar = areas[pbase + n];
    const float x  = __cosf(th);
    const float s  = sqrtf(fmaxf(1.0f - x * x, 0.0f));
    float sn, cn;
    __sincosf((float)m * ph, &sn, &cn);
    cmsm[w][lane] = make_float2(cn * ar, sn * ar);

    float sp = 1.0f;                           // s^m (m is wave-uniform)
    for (int i = 0; i < m; ++i) sp *= s;
    float p_prev = seedc * sp;                 // Nbar(m,m)
    As16[0 * 32 + lane] = (_Float16)p_prev;
    if (m < 32) {
      float p_cur = c1 * x * p_prev;           // Nbar(m+1,m)
      As16[1 * 32 + lane] = (_Float16)p_cur;
      for (int l = m + 2; l <= 32; ++l) {
        const float p_next = cA[l] * x * p_cur - cB[l] * p_prev;
        As16[(l - m) * 32 + lane] = (_Float16)p_next;
        p_prev = p_cur; p_cur = p_next;
      }
    }
    __builtin_amdgcn_wave_barrier();  // DS is in-order per wave; block IR reorder

    // -------- phase 2: build B (lane = channel x K-half), do WMMAs --------
    const float* vp = values + (pbase + n0 + Kbase) * Cq + ch;
    __builtin_prefetch((const void*)(vp + 256 * Cq), 0, 0);  // next chunk

    V16U Bc, Bs;
#pragma unroll
    for (int j = 0; j < 8; ++j) {
      const float v0 = vp[(2 * j) * Cq];
      const float v1 = vp[(2 * j + 1) * Cq];
      const float2 q0 = cmsm[w][Kbase + 2 * j];
      const float2 q1 = cmsm[w][Kbase + 2 * j + 1];
      Bc.e[2 * j]     = (_Float16)(v0 * q0.x);
      Bc.e[2 * j + 1] = (_Float16)(v1 * q1.x);
      Bs.e[2 * j]     = (_Float16)(v0 * q0.y);
      Bs.e[2 * j + 1] = (_Float16)(v1 * q1.y);
    }

    {
      const _Float16* ra = As16 + (0 * 16 + r16) * 32 + K0h;
      v16h A = cat8(*(const v8h*)ra, *(const v8h*)(ra + 8));
      aC0 = __builtin_amdgcn_wmma_f32_16x16x32_f16(false, A, false, Bc.v,
                                                   (short)0, aC0, false, false);
      if (m) aS0 = __builtin_amdgcn_wmma_f32_16x16x32_f16(false, A, false, Bs.v,
                                                   (short)0, aS0, false, false);
    }
    if (T > 1) {
      const _Float16* ra = As16 + (1 * 16 + r16) * 32 + K0h;
      v16h A = cat8(*(const v8h*)ra, *(const v8h*)(ra + 8));
      aC1 = __builtin_amdgcn_wmma_f32_16x16x32_f16(false, A, false, Bc.v,
                                                   (short)0, aC1, false, false);
      if (m) aS1 = __builtin_amdgcn_wmma_f32_16x16x32_f16(false, A, false, Bs.v,
                                                   (short)0, aS1, false, false);
    }
    if (T > 2) {
      const _Float16* ra = As16 + (2 * 16 + r16) * 32 + K0h;
      v16h A = cat8(*(const v8h*)ra, *(const v8h*)(ra + 8));
      aC2 = __builtin_amdgcn_wmma_f32_16x16x32_f16(false, A, false, Bc.v,
                                                   (short)0, aC2, false, false);
      if (m) aS2 = __builtin_amdgcn_wmma_f32_16x16x32_f16(false, A, false, Bs.v,
                                                   (short)0, aS2, false, false);
    }
    __builtin_amdgcn_wave_barrier();
  }

  // ---- epilogue: spill f32 tiles, fixed-order cross-wave sum -> split partial ----
  const int rrow = (lane >> 4) * 8;   // C/D layout: upper half-wave = rows +8
#pragma unroll
  for (int j = 0; j < 8; ++j) {
    const int r = j + rrow;
    red[w][0 * 256 + r * 16 + ch] = aC0[j];
    red[w][1 * 256 + r * 16 + ch] = aC1[j];
    red[w][2 * 256 + r * 16 + ch] = aC2[j];
    red[w][3 * 256 + r * 16 + ch] = aS0[j];
    red[w][4 * 256 + r * 16 + ch] = aS1[j];
    red[w][5 * 256 + r * 16 + ch] = aS2[j];
  }
  __syncthreads();
  for (int e = tid; e < 1536; e += 256) {
    float sum = 0.0f;
#pragma unroll
    for (int ww = 0; ww < 8; ++ww) sum += red[ww][e];   // fixed order
    ws[(size_t)wg * 1536 + e] = sum;
  }
}

// Sum the NSPLIT split-K partials (fixed order) and scatter with the closed-form
// output permutation: real(l,0)->l^2, real(l,m)->l^2+2m-1, imag(l,m)->l^2+2m.
__global__ __launch_bounds__(256)
void sphft_reduce(const float* __restrict__ ws, float* __restrict__ out) {
  const int bm = blockIdx.x;        // b*33 + m
  const int m  = bm % L1q;
  const int b  = bm / L1q;
  for (int e = threadIdx.x; e < 1536; e += 256) {
    float sum = 0.0f;
#pragma unroll
    for (int sp = 0; sp < NSPLIT; ++sp)
      sum += ws[((size_t)bm * NSPLIT + sp) * 1536 + e];
    const int  kt    = e >> 8;       // 0..5
    const int  t     = kt % 3;
    const bool isSin = kt >= 3;
    const int  idx   = e & 255;
    const int  r     = idx >> 4;
    const int  c     = idx & 15;
    const int  l     = m + t * 16 + r;
    if (l <= 32 && !(isSin && m == 0)) {
      const int row = isSin ? (l * l + 2 * m)
                            : (l * l + (m ? 2 * m - 1 : 0));
      out[((size_t)b * 1089 + row) * Cq + c] = isSin ? -sum : sum;
    }
  }
}

extern "C" void kernel_launch(void* const* d_in, const int* in_sizes, int n_in,
                              void* d_out, int out_size, void* d_ws, size_t ws_size,
                              hipStream_t stream) {
  (void)in_sizes; (void)n_in; (void)out_size; (void)ws_size;
  const float* theta  = (const float*)d_in[0];
  const float* phi    = (const float*)d_in[1];
  const float* areas  = (const float*)d_in[2];
  const float* values = (const float*)d_in[3];
  // d_in[4] = lmax (compile-time constant 32 here)
  float* ws  = (float*)d_ws;     // B*33*NSPLIT*1536 floats = ~3.2 MB
  float* out = (float*)d_out;

  sphft_main<<<dim3(Bq * L1q * NSPLIT), dim3(256), 0, stream>>>(
      theta, phi, areas, values, ws);
  sphft_reduce<<<dim3(Bq * L1q), dim3(256), 0, stream>>>(ws, out);
}
